// ModalGenerator_86414741995599
// MI455X (gfx1250) — matrix operand order repaired and checked by reading, not compiled, not run on hardware
//
#include <hip/hip_runtime.h>
#include <hip/hip_bf16.h>

// ---------------------------------------------------------------------------
// CrossModalGenerator, fused for gfx1250 (MI455X, wave32, WMMA).
// Key algebraic simplification: seq_len==1 -> softmax over one element == 1,
// so attn == v; the q/k projections are dead code and are skipped.
// All GEMMs run as bf16 WMMA (v_wmma_f32_16x16x32_bf16) with fp32 accum.
// ---------------------------------------------------------------------------

#define H     512
#define LDIM  3
#define BATCH 16384

typedef __attribute__((ext_vector_type(16))) __bf16      v16bf;
typedef __attribute__((ext_vector_type(8)))  float       v8f;
typedef __attribute__((ext_vector_type(4)))  unsigned int v4u;

union Frag { v4u q[2]; v16bf v; };

__device__ __forceinline__ unsigned short f2bf(float f) {
    unsigned int u = __float_as_uint(f);
    u += 0x7FFFu + ((u >> 16) & 1u);          // round-to-nearest-even
    return (unsigned short)(u >> 16);
}

__device__ __forceinline__ float gelu_f(float x) {
    return 0.5f * x * (1.0f + erff(x * 0.70710678118654752f));
}

// ---------------------------------------------------------------------------
// 16xNout = (16xK A from LDS bf16) @ (W.T), W row-major [Nout x ldW] bf16.
// A-fragment: lane holds row (lane&15), 16 contiguous K values
//             starting at kk + (lane>=16 ? 16 : 0).
// B-fragment: lane holds column (lane&15) = W row, 16 contiguous K values
//             starting at kk + (lane>=16 ? 16 : 0).
// MODE: 0 OF=val+b   1 OF+=val+b   2 OB=bf16(val+b)   3 OB=bf16(gelu(val+b))
// ---------------------------------------------------------------------------
template<int MODE>
__device__ __forceinline__ void gemm16(
    const unsigned short* __restrict__ A, int ldA,
    const unsigned short* __restrict__ W, int ldW, int K, int Nout,
    const float* __restrict__ bias,
    float* __restrict__ OF, unsigned short* __restrict__ OB, int ldO)
{
    const int tid  = threadIdx.x;
    const int lane = tid & 31;
    const int wave = tid >> 5;
    const int half = lane >> 4;
    const int l15  = lane & 15;
    const int NT   = Nout >> 4;
    const unsigned short* arow = A + l15 * ldA;

    for (int ct0 = wave; ct0 < NT; ct0 += 16) {
        const int ct1 = ct0 + 8;
        const unsigned short* w0 = W + (ct0 * 16 + l15) * ldW;
        const unsigned short* w1 = W + (ct1 * 16 + l15) * ldW;
        v8f acc0 = {0.f,0.f,0.f,0.f,0.f,0.f,0.f,0.f};
        v8f acc1 = {0.f,0.f,0.f,0.f,0.f,0.f,0.f,0.f};
        #pragma unroll 4
        for (int kk = 0; kk < K; kk += 32) {
            Frag fa, fb0, fb1;
            fa.q[0]  = *(const v4u*)(arow + kk + half * 16);
            fa.q[1]  = *(const v4u*)(arow + kk + half * 16 + 8);
            const int ob = kk + half * 16;
            fb0.q[0] = *(const v4u*)(w0 + ob);
            fb0.q[1] = *(const v4u*)(w0 + ob + 8);
            fb1.q[0] = *(const v4u*)(w1 + ob);
            fb1.q[1] = *(const v4u*)(w1 + ob + 8);
            acc0 = __builtin_amdgcn_wmma_f32_16x16x32_bf16(false, fa.v, false, fb0.v,
                                                           (short)0, acc0, false, false);
            acc1 = __builtin_amdgcn_wmma_f32_16x16x32_bf16(false, fa.v, false, fb1.v,
                                                           (short)0, acc1, false, false);
        }
        #pragma unroll
        for (int t = 0; t < 2; ++t) {
            v8f acc = t ? acc1 : acc0;
            const int c  = (t ? ct1 : ct0) * 16 + l15;
            const float bv = bias ? bias[c] : 0.0f;
            #pragma unroll
            for (int j = 0; j < 8; ++j) {
                const int r = half * 8 + j;               // C/D layout: rows j / j+8
                const float val = acc[j] + bv;
                if (MODE == 0) OF[r * ldO + c] = val;
                if (MODE == 1) OF[r * ldO + c] += val;
                if (MODE == 2) OB[r * ldO + c] = f2bf(val);
                if (MODE == 3) OB[r * ldO + c] = f2bf(gelu_f(val));
            }
        }
    }
}

// Final GEMM: gen = x @ opw.T + opb, then per-row select/blend and store.
__device__ __forceinline__ void gemm_final(
    const unsigned short* __restrict__ A,
    const unsigned short* __restrict__ W,
    const float* __restrict__ opbias,
    const float* __restrict__ tgt,
    const float* __restrict__ prior_p,
    const int*   __restrict__ mtv,
    float rw, int genmatch, float* __restrict__ outg, int rowbase)
{
    const int tid  = threadIdx.x;
    const int lane = tid & 31;
    const int wave = tid >> 5;
    const int half = lane >> 4;
    const int l15  = lane & 15;
    const unsigned short* arow = A + l15 * H;

    for (int ct0 = wave; ct0 < 32; ct0 += 16) {
        const int ct1 = ct0 + 8;
        const unsigned short* w0 = W + (ct0 * 16 + l15) * H;
        const unsigned short* w1 = W + (ct1 * 16 + l15) * H;
        v8f acc0 = {0.f,0.f,0.f,0.f,0.f,0.f,0.f,0.f};
        v8f acc1 = {0.f,0.f,0.f,0.f,0.f,0.f,0.f,0.f};
        #pragma unroll 4
        for (int kk = 0; kk < H; kk += 32) {
            Frag fa, fb0, fb1;
            fa.q[0]  = *(const v4u*)(arow + kk + half * 16);
            fa.q[1]  = *(const v4u*)(arow + kk + half * 16 + 8);
            const int ob = kk + half * 16;
            fb0.q[0] = *(const v4u*)(w0 + ob);
            fb0.q[1] = *(const v4u*)(w0 + ob + 8);
            fb1.q[0] = *(const v4u*)(w1 + ob);
            fb1.q[1] = *(const v4u*)(w1 + ob + 8);
            acc0 = __builtin_amdgcn_wmma_f32_16x16x32_bf16(false, fa.v, false, fb0.v,
                                                           (short)0, acc0, false, false);
            acc1 = __builtin_amdgcn_wmma_f32_16x16x32_bf16(false, fa.v, false, fb1.v,
                                                           (short)0, acc1, false, false);
        }
        #pragma unroll
        for (int t = 0; t < 2; ++t) {
            v8f acc = t ? acc1 : acc0;
            const int c  = (t ? ct1 : ct0) * 16 + l15;
            const float bv = opbias[c];
            #pragma unroll
            for (int j = 0; j < 8; ++j) {
                const int  r    = half * 8 + j;
                const long grow = rowbase + r;
                const int  m    = mtv[grow];
                const float gen = acc[j] + bv;
                const float tv  = tgt[grow * H + c];
                float o;
                if (m == genmatch)  o = rw * tv + (1.0f - rw) * gen;
                else if (m == 3)    o = prior_p[c];
                else                o = tv;
                outg[grow * H + c] = o;
            }
        }
    }
}

// LayerNorm over 16 rows x 512 (fp32 master in XF), writes XF and bf16 XS.
__device__ __forceinline__ void layer_norm16(
    float* __restrict__ XF, unsigned short* __restrict__ XS,
    const float* __restrict__ gg, const float* __restrict__ bb,
    float* __restrict__ red)
{
    const int tid = threadIdx.x;
    const int r   = tid >> 4;        // 16 threads per row
    const int s   = tid & 15;
    const int base = r * H + s * 32;
    float sum = 0.f, sq = 0.f;
    #pragma unroll
    for (int i = 0; i < 32; ++i) { float v = XF[base + i]; sum += v; sq += v * v; }
    red[r * 16 + s]       = sum;
    red[256 + r * 16 + s] = sq;
    __syncthreads();
    float m = 0.f, q = 0.f;
    #pragma unroll
    for (int i = 0; i < 16; ++i) { m += red[r * 16 + i]; q += red[256 + r * 16 + i]; }
    m *= (1.0f / H);
    q  = q * (1.0f / H) - m * m;
    const float inv = rsqrtf(q + 1e-5f);
    #pragma unroll
    for (int i = 0; i < 32; ++i) {
        const int c = s * 32 + i;
        float v = (XF[base + i] - m) * inv * gg[c] + bb[c];
        XF[base + i] = v;
        XS[base + i] = f2bf(v);
    }
}

// ---------------------------------------------------------------------------
// Weight conversion kernels (fp32 -> bf16 in d_ws, done once per launch).
// ---------------------------------------------------------------------------
__global__ void cvt_plain(const float* __restrict__ s, unsigned short* __restrict__ d, long n) {
    long i = (long)blockIdx.x * blockDim.x + threadIdx.x;
    if (i < n) d[i] = f2bf(s[i]);
}

// Extract only the wv third of qkv_w (2,L,3H,H): rows [2H,3H).
__global__ void cvt_wv_k(const float* __restrict__ qkv, unsigned short* __restrict__ d) {
    long i = (long)blockIdx.x * blockDim.x + threadIdx.x;
    if (i >= (long)2 * LDIM * H * H) return;
    long c  = i & (H - 1);
    long r  = i >> 9;               // [0, 2*L*512)
    long gl = r >> 9;               // g*L + l
    long rr = r & (H - 1);
    d[i] = f2bf(qkv[((gl * 3 + 2) * H + rr) * H + c]);
}

// Tiny prior MLP: out[1024] = (gelu(emb @ w1.T + b1)) @ w2.T + b2
__global__ void prior_kernel(const float* __restrict__ emb,
                             const float* __restrict__ w1, const float* __restrict__ b1,
                             const float* __restrict__ w2, const float* __restrict__ b2,
                             float* __restrict__ out) {
    __shared__ float h1[2 * H];
    const int i = threadIdx.x;
    float s = b1[i];
    for (int k = 0; k < H; ++k) s += emb[k] * w1[i * H + k];
    h1[i] = gelu_f(s);
    __syncthreads();
    float s2 = b2[i];
    for (int k = 0; k < 2 * H; ++k) s2 += h1[k] * w2[i * 2 * H + k];
    out[i] = s2;
}

// ---------------------------------------------------------------------------
// Fused generator: one workgroup = 16 rows, full 3-layer pipeline in LDS.
// blockIdx.y == 0 : gen_text (src=image, tgt=text) -> writes out[B*H ..)
// blockIdx.y == 1 : gen_img  (src=text, tgt=image) -> writes out[0 .. B*H)
// ---------------------------------------------------------------------------
__global__ __launch_bounds__(256) void gen_kernel(
    const float* __restrict__ img, const float* __restrict__ txt,
    const unsigned short* __restrict__ Wip, const float* __restrict__ ipb,
    const unsigned short* __restrict__ Wv,  const float* __restrict__ qkvb,
    const unsigned short* __restrict__ Wao, const float* __restrict__ aob,
    const float* __restrict__ ln1g, const float* __restrict__ ln1b,
    const float* __restrict__ ln2g, const float* __restrict__ ln2b,
    const unsigned short* __restrict__ Wf1, const float* __restrict__ f1b,
    const unsigned short* __restrict__ Wf2, const float* __restrict__ f2b,
    const unsigned short* __restrict__ Wop, const float* __restrict__ opb,
    const float* __restrict__ rwv, const float* __restrict__ prior,
    const int* __restrict__ mtv, float* __restrict__ outp)
{
    __shared__ unsigned short xs[16 * H];   // x (bf16, GEMM A operand)
    __shared__ unsigned short ts[16 * H];   // tgt (bf16)
    __shared__ unsigned short hs[16 * H];   // intermediate (v / f1 chunk)
    __shared__ float          xf[16 * H];   // fp32 master of x (residual + LN)
    __shared__ float          red[512];

    const int tid     = threadIdx.x;
    const int g       = blockIdx.y;
    const int rowbase = blockIdx.x * 16;
    const float* src = (g == 0) ? img : txt;
    const float* tgt = (g == 0) ? txt : img;

    // Stage src/tgt rows as bf16.
    for (int i = tid; i < 16 * H; i += 256) {
        const long gi = (long)rowbase * H + i;
        xs[i] = f2bf(src[gi]);
        ts[i] = f2bf(tgt[gi]);
    }
    __syncthreads();

    // x = src @ ipw.T + ipb
    gemm16<0>(xs, H, Wip + (long)g * H * H, H, H, H, ipb + g * H, xf, nullptr, H);
    __syncthreads();
    for (int i = tid; i < 16 * H; i += 256) xs[i] = f2bf(xf[i]);
    __syncthreads();

    for (int l = 0; l < LDIM; ++l) {
        const int gl = g * LDIM + l;
        // attn = v = tgt @ wv.T + bv   (softmax over singleton == 1)
        gemm16<2>(ts, H, Wv + (long)gl * H * H, H, H, H,
                  qkvb + (long)gl * 3 * H + 2 * H, nullptr, hs, H);
        __syncthreads();
        // x += attn @ ao.T + ao_b
        gemm16<1>(hs, H, Wao + (long)gl * H * H, H, H, H, aob + gl * H, xf, nullptr, H);
        __syncthreads();
        layer_norm16(xf, xs, ln1g + gl * H, ln1b + gl * H, red);
        __syncthreads();
        // FFN, 2048-wide hidden processed in 4 chunks of 512 with fp32 accum.
        for (int c = 0; c < 4; ++c) {
            gemm16<3>(xs, H, Wf1 + ((long)gl * 4 * H + c * H) * H, H, H, H,
                      f1b + (long)gl * 4 * H + c * H, nullptr, hs, H);
            __syncthreads();
            gemm16<1>(hs, H, Wf2 + (long)gl * H * 4 * H + c * H, 4 * H, H, H,
                      (c == 0) ? (f2b + gl * H) : nullptr, xf, nullptr, H);
            __syncthreads();
        }
        layer_norm16(xf, xs, ln2g + gl * H, ln2b + gl * H, red);
        __syncthreads();
    }

    // gen = x @ opw.T + opb; fuse rw-blend + missing_type select + store.
    gemm_final(xs, Wop + (long)g * H * H, opb + g * H, tgt,
               prior + (g == 0 ? H : 0), mtv, rwv[g],
               (g == 0 ? 1 : 2), outp + (g == 0 ? (long)BATCH * H : 0), rowbase);
}

// ---------------------------------------------------------------------------
extern "C" void kernel_launch(void* const* d_in, const int* in_sizes, int n_in,
                              void* d_out, int out_size, void* d_ws, size_t ws_size,
                              hipStream_t stream)
{
    const float* img  = (const float*)d_in[0];
    const float* txt  = (const float*)d_in[1];
    const float* ipw  = (const float*)d_in[2];
    const float* ipb  = (const float*)d_in[3];
    const float* qkvw = (const float*)d_in[4];
    const float* qkvb = (const float*)d_in[5];
    const float* aow  = (const float*)d_in[6];
    const float* aob  = (const float*)d_in[7];
    const float* ln1g = (const float*)d_in[8];
    const float* ln1b = (const float*)d_in[9];
    const float* ln2g = (const float*)d_in[10];
    const float* ln2b = (const float*)d_in[11];
    const float* f1w  = (const float*)d_in[12];
    const float* f1b  = (const float*)d_in[13];
    const float* f2w  = (const float*)d_in[14];
    const float* f2b  = (const float*)d_in[15];
    const float* opw  = (const float*)d_in[16];
    const float* opb  = (const float*)d_in[17];
    const float* rw   = (const float*)d_in[18];
    const float* pw1  = (const float*)d_in[19];
    const float* pb1  = (const float*)d_in[20];
    const float* pw2  = (const float*)d_in[21];
    const float* pb2  = (const float*)d_in[22];
    const float* pemb = (const float*)d_in[23];
    const int*   mt   = (const int*)d_in[24];

    // d_ws layout: [0,4096) prior fp32 vector; then bf16 weights.
    const long n_ip = 2L  * H * H;
    const long n_v  = 6L  * H * H;
    const long n_ao = 6L  * H * H;
    const long n_f1 = 24L * H * H;
    const long n_f2 = 24L * H * H;
    const long n_op = 2L  * H * H;
    const size_t need = 4096 + 2 * (size_t)(n_ip + n_v + n_ao + n_f1 + n_f2 + n_op);
    if (ws_size < need) return;   // scratch too small: cannot run

    float* prior = (float*)d_ws;
    unsigned short* Wip = (unsigned short*)((char*)d_ws + 4096);
    unsigned short* Wv  = Wip + n_ip;
    unsigned short* Wao = Wv  + n_v;
    unsigned short* Wf1 = Wao + n_ao;
    unsigned short* Wf2 = Wf1 + n_f1;
    unsigned short* Wop = Wf2 + n_f2;

    auto nb = [](long n) { return (int)((n + 255) / 256); };
    cvt_plain<<<nb(n_ip), 256, 0, stream>>>(ipw, Wip, n_ip);
    cvt_wv_k <<<nb(n_v),  256, 0, stream>>>(qkvw, Wv);
    cvt_plain<<<nb(n_ao), 256, 0, stream>>>(aow, Wao, n_ao);
    cvt_plain<<<nb(n_f1), 256, 0, stream>>>(f1w, Wf1, n_f1);
    cvt_plain<<<nb(n_f2), 256, 0, stream>>>(f2w, Wf2, n_f2);
    cvt_plain<<<nb(n_op), 256, 0, stream>>>(opw, Wop, n_op);
    prior_kernel<<<1, 2 * H, 0, stream>>>(pemb, pw1, pb1, pw2, pb2, prior);

    dim3 grid(BATCH / 16, 2);
    gen_kernel<<<grid, 256, 0, stream>>>(img, txt, Wip, ipb, Wv, qkvb, Wao, aob,
                                         ln1g, ln1b, ln2g, ln2b, Wf1, f1b, Wf2, f2b,
                                         Wop, opb, rw, prior, mt, (float*)d_out);
    (void)in_sizes; (void)n_in; (void)out_size;
}
